// construct_34522947125930
// MI455X (gfx1250) — matrix operand order repaired and checked by reading, not compiled, not run on hardware
//
#include <hip/hip_runtime.h>
#include <math.h>

typedef __bf16 bf16_t;
typedef __attribute__((ext_vector_type(16))) __bf16 v16bf;
typedef __attribute__((ext_vector_type(8)))  __bf16 v8bf;
typedef __attribute__((ext_vector_type(8)))  float   v8f;

// Problem constants
#define BB   32
#define TS   512
#define DIN  128
#define DOUT 128
#define DD   1024
#define GG   4096   // 4*D
#define MM   (BB*TS)

// ---------------- elementwise f32 -> bf16 ----------------
__global__ void k_convert_bf16(const float* __restrict__ src, bf16_t* __restrict__ dst, int n) {
  int i = blockIdx.x * blockDim.x + threadIdx.x;
  if (i < n) dst[i] = (bf16_t)src[i];
}

// ---------------- transpose+convert: src[K,N] f32 -> dst[N,K] bf16 ----------------
__global__ void k_transpose_bf16(const float* __restrict__ src, bf16_t* __restrict__ dst, int K, int N) {
  int i = blockIdx.x * blockDim.x + threadIdx.x;
  if (i >= K * N) return;
  int n = i / K, k = i - n * K;
  dst[(size_t)n * K + k] = (bf16_t)src[(size_t)k * N + n];
}

// ---------------- init recurrent state ----------------
__global__ void k_init_hc(const float* __restrict__ h0, const float* __restrict__ c0,
                          float* __restrict__ h, float* __restrict__ c,
                          bf16_t* __restrict__ hbf, int n) {
  int i = blockIdx.x * blockDim.x + threadIdx.x;
  if (i < n) { float hv = h0[i]; h[i] = hv; c[i] = c0[i]; hbf[i] = (bf16_t)hv; }
}

// ---- WMMA bf16 GEMM: C = (Cin + bias) + A[M,K] @ Wt[N,K]^T ----
// One wave computes a 32x16 strip of C (2 m-tiles sharing one B fragment):
// halves weight traffic and interleaves two independent WMMA accumulator
// chains to hide the WMMA->WMMA RAW latency. M must be a multiple of 32.
__global__ void k_wmma_gemm(const bf16_t* __restrict__ A, long lda,
                            const bf16_t* __restrict__ Wt, long ldw,
                            const float*  __restrict__ Cin, long ldcin,
                            const float*  __restrict__ bias,
                            float*  __restrict__ Cout, long ldcout,
                            bf16_t* __restrict__ Cbf,  long ldcbf,
                            int M, int N, int K) {
  int wave = (blockIdx.x * blockDim.x + threadIdx.x) >> 5;
  int lane = threadIdx.x & 31;
  int ntiles = N >> 4;
  int mstrips = M >> 5;               // 32-row strips
  if (wave >= mstrips * ntiles) return;
  int ms = wave / ntiles, nt = wave % ntiles;
  int row = lane & 15;        // A row in tile / B column in tile
  int hi  = lane >> 4;        // lane-half selector
  int m0 = ms * 32, n0 = nt * 16;

  // A fragments (16x32 bf16 each): lanes 0-15: K 0-7 & 16-23; lanes 16-31: K 8-15 & 24-31
  const bf16_t* ap0 = A + (size_t)(m0 + row) * lda + hi * 8;
  const bf16_t* ap1 = ap0 + (size_t)16 * lda;
  // Shared B fragment (32x16 bf16) from Wt[N,K]: lanes 0-15: K 0-7 & 16-23; lanes 16-31: K 8-15 & 24-31
  const bf16_t* bp  = Wt + (size_t)(n0 + row) * ldw + hi * 8;

  v8f acc0, acc1;
  float bval = bias ? bias[n0 + row] : 0.0f;
  if (Cin) {
    #pragma unroll
    for (int r = 0; r < 8; ++r) {
      acc0[r] = Cin[(size_t)(m0 + r + hi * 8) * ldcin + n0 + row] + bval;
      acc1[r] = Cin[(size_t)(m0 + 16 + r + hi * 8) * ldcin + n0 + row] + bval;
    }
  } else {
    #pragma unroll
    for (int r = 0; r < 8; ++r) { acc0[r] = bval; acc1[r] = bval; }
  }

  for (int k0 = 0; k0 < K; k0 += 32) {
    v8bf a00 = *(const v8bf*)(ap0 + k0);
    v8bf a01 = *(const v8bf*)(ap0 + k0 + 16);
    v8bf a10 = *(const v8bf*)(ap1 + k0);
    v8bf a11 = *(const v8bf*)(ap1 + k0 + 16);
    v8bf b0  = *(const v8bf*)(bp + k0);
    v8bf b1  = *(const v8bf*)(bp + k0 + 16);
    v16bf af0, af1, bfr;
    #pragma unroll
    for (int i = 0; i < 8; ++i) {
      af0[i] = a00[i]; af0[i + 8] = a01[i];
      af1[i] = a10[i]; af1[i + 8] = a11[i];
      bfr[i] = b0[i];  bfr[i + 8] = b1[i];
    }
    acc0 = __builtin_amdgcn_wmma_f32_16x16x32_bf16(false, af0, false, bfr, (short)0, acc0, false, false);
    acc1 = __builtin_amdgcn_wmma_f32_16x16x32_bf16(false, af1, false, bfr, (short)0, acc1, false, false);
  }

  if (Cout) {
    #pragma unroll
    for (int r = 0; r < 8; ++r) {
      Cout[(size_t)(m0 + r + hi * 8) * ldcout + n0 + row] = acc0[r];
      Cout[(size_t)(m0 + 16 + r + hi * 8) * ldcout + n0 + row] = acc1[r];
    }
  }
  if (Cbf) {
    #pragma unroll
    for (int r = 0; r < 8; ++r) {
      Cbf[(size_t)(m0 + r + hi * 8) * ldcbf + n0 + row] = (bf16_t)acc0[r];
      Cbf[(size_t)(m0 + 16 + r + hi * 8) * ldcbf + n0 + row] = (bf16_t)acc1[r];
    }
  }
}

// ---------------- LSTM gates + state update ----------------
__device__ __forceinline__ float sigm(float x) { return 1.0f / (1.0f + __expf(-x)); }

__global__ void k_lstm_point(const float* __restrict__ z,  // [B, 4D]
                             float* __restrict__ h, float* __restrict__ c,
                             bf16_t* __restrict__ hbf,
                             float* __restrict__ seqf, long seqf_ld,     // nullable
                             bf16_t* __restrict__ seqbf, long seqbf_ld)  // nullable
{
  int i = blockIdx.x * blockDim.x + threadIdx.x;
  if (i >= BB * DD) return;
  int b = i >> 10, d = i & (DD - 1);
  const float* zb = z + (size_t)b * GG;
  float iv = sigm(zb[d]);
  float fv = sigm(zb[d + DD]);
  float gv = tanhf(zb[d + 2 * DD]);
  float ov = sigm(zb[d + 3 * DD]);
  float cv = fv * c[i] + iv * gv;
  float hv = ov * tanhf(cv);
  c[i] = cv; h[i] = hv; hbf[i] = (bf16_t)hv;
  if (seqf)  seqf [(size_t)b * seqf_ld  + d] = hv;
  if (seqbf) seqbf[(size_t)b * seqbf_ld + d] = (bf16_t)hv;
}

// ---------------- attention: scores -> softmax -> context (per batch block) ----------------
__global__ void k_attn_ctx(const float* __restrict__ enc, // [B*T, D], row = b*TS + t
                           const float* __restrict__ h,   // [B, D]
                           bf16_t* __restrict__ ctxbf)    // [B, D]
{
  __shared__ float sc[TS];
  __shared__ float red[256];
  int b = blockIdx.x;
  int tid = threadIdx.x;
  const float* eb = enc + (size_t)b * TS * DD;
  const float* hb = h + (size_t)b * DD;

  for (int tt = tid; tt < TS; tt += 256) {
    const float* er = eb + (size_t)tt * DD;
    float s = 0.f;
    for (int d = 0; d < DD; ++d) s += er[d] * hb[d];
    sc[tt] = s;
  }
  __syncthreads();
  float m = -INFINITY;
  for (int tt = tid; tt < TS; tt += 256) m = fmaxf(m, sc[tt]);
  red[tid] = m; __syncthreads();
  for (int s = 128; s > 0; s >>= 1) { if (tid < s) red[tid] = fmaxf(red[tid], red[tid + s]); __syncthreads(); }
  m = red[0]; __syncthreads();
  float psum = 0.f;
  for (int tt = tid; tt < TS; tt += 256) { float e = __expf(sc[tt] - m); sc[tt] = e; psum += e; }
  red[tid] = psum; __syncthreads();
  for (int s = 128; s > 0; s >>= 1) { if (tid < s) red[tid] += red[tid + s]; __syncthreads(); }
  float inv = 1.0f / red[0];
  __syncthreads();
  for (int d = tid; d < DD; d += 256) {
    float acc = 0.f;
    for (int tt = 0; tt < TS; ++tt) acc += sc[tt] * eb[(size_t)tt * DD + d];
    ctxbf[(size_t)b * DD + d] = (bf16_t)(acc * inv);
  }
}

// ---------------- host-side launch helpers ----------------
static inline void gemm(hipStream_t s, const bf16_t* A, long lda, const bf16_t* Wt, long ldw,
                        const float* Cin, long ldcin, const float* bias,
                        float* Cout, long ldcout, bf16_t* Cbf, long ldcbf,
                        int M, int N, int K) {
  int tiles = (M / 32) * (N / 16);    // 32x16 strips
  int blocks = (tiles + 3) / 4;
  k_wmma_gemm<<<blocks, 128, 0, s>>>(A, lda, Wt, ldw, Cin, ldcin, bias, Cout, ldcout, Cbf, ldcbf, M, N, K);
}

extern "C" void kernel_launch(void* const* d_in, const int* in_sizes, int n_in,
                              void* d_out, int out_size, void* d_ws, size_t ws_size,
                              hipStream_t stream) {
  const float* inputs  = (const float*)d_in[0];
  const float* outputs = (const float*)d_in[1];
  const float* h0      = (const float*)d_in[2];
  const float* c0      = (const float*)d_in[3];
  const float* Wi      = (const float*)d_in[4];
  const float* bi      = (const float*)d_in[5];
  const float* enc_k   = (const float*)d_in[6];
  const float* enc_rk  = (const float*)d_in[7];
  const float* enc_b   = (const float*)d_in[8];
  const float* Wo      = (const float*)d_in[9];
  const float* bo      = (const float*)d_in[10];
  const float* stk_k   = (const float*)d_in[11];
  const float* stk_rk  = (const float*)d_in[12];
  const float* stk_b   = (const float*)d_in[13];
  const float* att_k   = (const float*)d_in[14];
  const float* att_rk  = (const float*)d_in[15];
  const float* att_b   = (const float*)d_in[16];
  const float* Wf      = (const float*)d_in[17];
  const float* bf_     = (const float*)d_in[18];
  float* out           = (float*)d_out;

  // ---- workspace layout ----
  char* ws = (char*)d_ws;
  size_t off = 0;
  auto take = [&](size_t bytes) -> char* {
    char* p = ws + off;
    off = (off + bytes + 255) & ~(size_t)255;
    return p;
  };
  float*  xz      = (float*)take((size_t)MM * GG * 4);       // hoisted input-side z
  float*  encf    = (float*)take((size_t)MM * DD * 4);       // encoder hidden seq (f32)
  float*  z       = (float*)take((size_t)BB * GG * 4);       // per-step gates
  float*  h       = (float*)take((size_t)BB * DD * 4);
  float*  c       = (float*)take((size_t)BB * DD * 4);
  bf16_t* inp_bf  = (bf16_t*)take((size_t)MM * DIN * 2);
  bf16_t* outp_bf = (bf16_t*)take((size_t)MM * DOUT * 2);
  bf16_t* x_bf    = (bf16_t*)take((size_t)MM * DD * 2);      // layer-input / layer-output seq (bf16)
  bf16_t* h_bf    = (bf16_t*)take((size_t)BB * DD * 2);
  bf16_t* ctx_bf  = (bf16_t*)take((size_t)BB * DD * 2);
  bf16_t* WiT     = (bf16_t*)take((size_t)DD * DIN * 2);     // [1024,128]
  bf16_t* WoT     = (bf16_t*)take((size_t)DD * DOUT * 2);
  bf16_t* enc_kT  = (bf16_t*)take((size_t)GG * DD * 2);      // [4096,1024]
  bf16_t* enc_rkT = (bf16_t*)take((size_t)GG * DD * 2);
  bf16_t* stk_kT  = (bf16_t*)take((size_t)2 * GG * DD * 2);
  bf16_t* stk_rkT = (bf16_t*)take((size_t)2 * GG * DD * 2);
  bf16_t* att_kT  = (bf16_t*)take((size_t)GG * 2 * DD * 2);  // [4096,2048]
  bf16_t* att_rkT = (bf16_t*)take((size_t)GG * DD * 2);
  bf16_t* WfT     = (bf16_t*)take((size_t)DOUT * DD * 2);    // [128,1024]
  (void)ws_size; (void)in_sizes; (void)n_in; (void)out_size;

  // ---- one-time conversions / weight transposes ----
  {
    int n;
    n = MM * DIN;  k_convert_bf16<<<(n + 255) / 256, 256, 0, stream>>>(inputs,  inp_bf,  n);
    n = MM * DOUT; k_convert_bf16<<<(n + 255) / 256, 256, 0, stream>>>(outputs, outp_bf, n);
    n = DIN * DD;      k_transpose_bf16<<<(n + 255) / 256, 256, 0, stream>>>(Wi, WiT, DIN, DD);
    n = DOUT * DD;     k_transpose_bf16<<<(n + 255) / 256, 256, 0, stream>>>(Wo, WoT, DOUT, DD);
    n = DD * GG;       k_transpose_bf16<<<(n + 255) / 256, 256, 0, stream>>>(enc_k,  enc_kT,  DD, GG);
    n = DD * GG;       k_transpose_bf16<<<(n + 255) / 256, 256, 0, stream>>>(enc_rk, enc_rkT, DD, GG);
    for (int l = 0; l < 2; ++l) {
      n = DD * GG;
      k_transpose_bf16<<<(n + 255) / 256, 256, 0, stream>>>(stk_k  + (size_t)l * DD * GG, stk_kT  + (size_t)l * GG * DD, DD, GG);
      k_transpose_bf16<<<(n + 255) / 256, 256, 0, stream>>>(stk_rk + (size_t)l * DD * GG, stk_rkT + (size_t)l * GG * DD, DD, GG);
    }
    n = 2 * DD * GG;   k_transpose_bf16<<<(n + 255) / 256, 256, 0, stream>>>(att_k, att_kT, 2 * DD, GG);
    n = DD * GG;       k_transpose_bf16<<<(n + 255) / 256, 256, 0, stream>>>(att_rk, att_rkT, DD, GG);
    n = DD * DOUT;     k_transpose_bf16<<<(n + 255) / 256, 256, 0, stream>>>(Wf, WfT, DD, DOUT);
  }

  const long seq_ld = (long)TS * DD;   // row stride (per batch) for [B*T, D] seq buffers
  const long xz_ld  = (long)TS * GG;   // per-batch row stride into xz at fixed t

  // ---- encoder branch: x_in = inputs@Wi + bi (bf16) ----
  gemm(stream, inp_bf, DIN, WiT, DIN, nullptr, 0, bi, nullptr, 0, x_bf, DD, MM, DD, DIN);
  // hoisted input-side gates: xz = x_in @ enc_k + enc_b
  gemm(stream, x_bf, DD, enc_kT, DD, nullptr, 0, enc_b, xz, GG, nullptr, 0, MM, GG, DD);
  k_init_hc<<<(BB * DD + 255) / 256, 256, 0, stream>>>(h0, c0, h, c, h_bf, BB * DD);
  for (int t = 0; t < TS; ++t) {
    gemm(stream, h_bf, DD, enc_rkT, DD, xz + (size_t)t * GG, xz_ld, nullptr, z, GG, nullptr, 0, BB, GG, DD);
    k_lstm_point<<<(BB * DD + 255) / 256, 256, 0, stream>>>(z, h, c, h_bf, encf + (size_t)t * DD, seq_ld, nullptr, 0);
  }

  // ---- decoder-input branch: y = outputs@Wo + bo ----
  gemm(stream, outp_bf, DOUT, WoT, DOUT, nullptr, 0, bo, nullptr, 0, x_bf, DD, MM, DD, DOUT);
  for (int l = 0; l < 2; ++l) {
    gemm(stream, x_bf, DD, stk_kT + (size_t)l * GG * DD, DD, nullptr, 0, stk_b + (size_t)l * GG, xz, GG, nullptr, 0, MM, GG, DD);
    k_init_hc<<<(BB * DD + 255) / 256, 256, 0, stream>>>(h0, c0, h, c, h_bf, BB * DD);
    for (int t = 0; t < TS; ++t) {
      gemm(stream, h_bf, DD, stk_rkT + (size_t)l * GG * DD, DD, xz + (size_t)t * GG, xz_ld, nullptr, z, GG, nullptr, 0, BB, GG, DD);
      k_lstm_point<<<(BB * DD + 255) / 256, 256, 0, stream>>>(z, h, c, h_bf, nullptr, 0, x_bf + (size_t)t * DD, seq_ld);
    }
  }

  // ---- attention decoder: xt-part hoisted, ctx-part per step ----
  gemm(stream, x_bf, DD, att_kT, 2 * DD, nullptr, 0, att_b, xz, GG, nullptr, 0, MM, GG, DD);
  k_init_hc<<<(BB * DD + 255) / 256, 256, 0, stream>>>(h0, c0, h, c, h_bf, BB * DD);
  for (int t = 0; t < TS; ++t) {
    k_attn_ctx<<<BB, 256, 0, stream>>>(encf, h, ctx_bf);
    gemm(stream, h_bf, DD, att_rkT, DD, xz + (size_t)t * GG, xz_ld, nullptr, z, GG, nullptr, 0, BB, GG, DD);
    gemm(stream, ctx_bf, DD, att_kT + DD, 2 * DD, z, GG, nullptr, z, GG, nullptr, 0, BB, GG, DD);
    k_lstm_point<<<(BB * DD + 255) / 256, 256, 0, stream>>>(z, h, c, h_bf, nullptr, 0, x_bf + (size_t)t * DD, seq_ld);
  }

  // ---- final projection: out = dec @ Wf + bf ----
  gemm(stream, x_bf, DD, WfT, DD, nullptr, 0, bf_, out, DOUT, nullptr, 0, MM, DOUT, DD);
}
